// RNN_50311246905429
// MI455X (gfx1250) — matrix operand order repaired and checked by reading, not compiled, not run on hardware
//
#include <hip/hip_runtime.h>
#include <hip/hip_bf16.h>
#include <stdint.h>

#define B_   64
#define T_   512
#define I_   512
#define H_   1024
#define G_   (3*H_)
#define O_   512
#define BT_  (B_*T_)
#define NBLK 32

typedef __attribute__((ext_vector_type(16))) __bf16 v16bf;
typedef __attribute__((ext_vector_type(8)))  float  v8f;
typedef __attribute__((ext_vector_type(4)))  unsigned int u32x4;
typedef __attribute__((ext_vector_type(8)))  int i32x8;
typedef __attribute__((ext_vector_type(4)))  int i32x4;

union Frag { u32x4 u[2]; v16bf v; };

// ---- CDNA5 async global->LDS copy (16B per lane), ASYNCcnt-tracked ----
__device__ __forceinline__ void async_copy16(unsigned lds_dst, const void* gsrc) {
  asm volatile("global_load_async_to_lds_b128 %0, %1, off"
               :: "v"(lds_dst), "v"(gsrc) : "memory");
}
__device__ __forceinline__ void async_wait0() {
  asm volatile("s_wait_asynccnt 0" ::: "memory");
}

// ---- CDNA5 Tensor Data Mover: load a 16-row x krow-col bf16 tile into LDS.
// Pad 4 DWORDs (16B) after every 256 DWORDs (1024B) -> conflict-free layout.
// D# encoding per CDNA5 ISA.
__device__ __forceinline__ void tdm_load_tile16(unsigned lds_base, const void* gsrc,
                                                int krow) {
  unsigned long long ga = (unsigned long long)(uintptr_t)gsrc;
  u32x4 g0;
  g0[0] = 1u;                                           // count=1 (valid D#)
  g0[1] = lds_base;                                     // lds_addr (bytes)
  g0[2] = (unsigned)(ga & 0xFFFFFFFFu);                 // global_addr[31:0]
  g0[3] = ((unsigned)(ga >> 32) & 0x1FFFFFFu) | (2u << 30);  // addr[56:32] | type=2
  i32x8 g1;
  g1[0] = (int)((1u << 16)      // data_size = 1 -> 2 bytes
              | (1u << 20)      // pad_enable
              | (7u << 22)      // pad_interval: 256 DWORDs
              | (3u << 25));    // pad_amount: 4 DWORDs (16B)
  g1[1] = (int)(((unsigned)krow & 0xFFFFu) << 16);      // tensor_dim0 lo16
  g1[2] = (int)(((unsigned)krow >> 16) | (16u << 16));  // tensor_dim0 hi16 | tensor_dim1=16
  g1[3] = (int)((unsigned)krow << 16);                  // tile_dim0 = krow
  g1[4] = 16;                                           // tile_dim1 = 16, tile_dim2 = 0
  g1[5] = krow;                                         // tensor_dim0_stride lo32
  g1[6] = 0;
  g1[7] = 0;
  i32x4 z4 = {};
#if __clang_major__ >= 23
  i32x8 z8 = {};
  __builtin_amdgcn_tensor_load_to_lds(g0, g1, z4, z4, z8, 0);
#else
  __builtin_amdgcn_tensor_load_to_lds(g0, g1, z4, z4, 0);
#endif
}

// B-matrix 32x16 bf16 fragment from a row-major (N x K) weight (B = W^T).
__device__ __forceinline__ v16bf frag_b(const __hip_bfloat16* __restrict__ wrow,
                                        int ldw, int n, int k0, int lane) {
  int kb = (lane >> 4) * 8;
  const __hip_bfloat16* p = wrow + (size_t)n * ldw + k0 + kb;
  Frag f;
  f.u[0] = *reinterpret_cast<const u32x4*>(p);
  f.u[1] = *reinterpret_cast<const u32x4*>(p + 16);
  return f.v;
}

// A fragment from LDS, simple padded-row layout (row stride astr elements).
__device__ __forceinline__ v16bf frag_a_lds(const __hip_bfloat16* lds,
                                            int astr, int k0, int lane) {
  int kb = (lane >> 4) * 8;
  const __hip_bfloat16* p = lds + (lane & 15) * astr + k0 + kb;
  Frag f;
  f.u[0] = *reinterpret_cast<const u32x4*>(p);
  f.u[1] = *reinterpret_cast<const u32x4*>(p + 16);
  return f.v;
}

// TDM-padded layout for 1024-elem rows: 512-elem segments with 8-elem pads.
__device__ __forceinline__ int hoff(int m, int k) {
  return ((m << 1) + (k >> 9)) * 520 + (k & 511);
}
__device__ __forceinline__ v16bf frag_a_hsh(const __hip_bfloat16* lds,
                                            int k0, int lane) {
  int kb = (lane >> 4) * 8;
  const __hip_bfloat16* p = lds + hoff(lane & 15, k0 + kb);
  Frag f;
  f.u[0] = *reinterpret_cast<const u32x4*>(p);
  f.u[1] = *reinterpret_cast<const u32x4*>(p + 16);
  return f.v;
}

__global__ void cast_f32_bf16(const float* __restrict__ src,
                              __hip_bfloat16* __restrict__ dst, int n) {
  int i = blockIdx.x * blockDim.x + threadIdx.x;
  int stride = gridDim.x * blockDim.x;
  for (; i < n; i += stride) dst[i] = __float2bfloat16(src[i]);
}

__global__ void init_scan(__hip_bfloat16* __restrict__ hstate,
                          unsigned* __restrict__ counter) {
  int i = blockIdx.x * blockDim.x + threadIdx.x;
  if (i < B_ * H_) hstate[i] = __float2bfloat16(0.0f);
  if (i == 0) *counter = 0u;
}

// C(M x N) = A(M x K, bf16) * W^T(N x K, bf16) + bias, fp32 out, ldc == N.
// Block tile 16(M) x 512(N); wave tile 16 x 64. A tile staged in LDS via TDM
// (K==512) or per-lane async copies (otherwise).
__global__ void gemm_bias_wmma(const __hip_bfloat16* __restrict__ A,
                               const __hip_bfloat16* __restrict__ W,
                               const float* __restrict__ bias,
                               float* __restrict__ C,
                               int M, int N, int K) {
  __shared__ __align__(16) __hip_bfloat16 ash[16 * 1032];
  const int lane = threadIdx.x & 31;
  const int wave = threadIdx.x >> 5;
  const int n0 = blockIdx.x * 512 + wave * 64;
  const int m0 = blockIdx.y * 16;
  const int nL = lane & 15;
  const int astr = K + 8;

  if (K == 512) {
    if (wave == 0) {
      unsigned lds_base = (unsigned)(uintptr_t)(&ash[0]);
      tdm_load_tile16(lds_base, A + (size_t)m0 * K, K);
      __builtin_amdgcn_s_wait_tensorcnt(0);
    }
  } else {
    const int cpr = K >> 3;
    const int chunks = 16 * cpr;
    for (int idx = threadIdx.x; idx < chunks; idx += 256) {
      int m = idx / cpr;
      int c8 = idx - m * cpr;
      unsigned dst = (unsigned)(uintptr_t)(&ash[m * astr + c8 * 8]);
      async_copy16(dst, A + (size_t)(m0 + m) * K + c8 * 8);
    }
    async_wait0();
  }
  __syncthreads();

  v8f acc[4];
#pragma unroll
  for (int j = 0; j < 4; ++j) acc[j] = (v8f){};

  for (int k0 = 0; k0 < K; k0 += 32) {
    v16bf a = frag_a_lds(ash, astr, k0, lane);
    v16bf bf[4];
#pragma unroll
    for (int j = 0; j < 4; ++j)
      bf[j] = frag_b(W, K, n0 + j * 16 + nL, k0, lane);
#pragma unroll
    for (int j = 0; j < 4; ++j)
      acc[j] = __builtin_amdgcn_wmma_f32_16x16x32_bf16(false, a, false, bf[j],
                                                       (short)0, acc[j], false, false);
  }

  const int mb = (lane >> 4) << 3;
#pragma unroll
  for (int j = 0; j < 4; ++j) {
    int n = n0 + j * 16 + nL;
    float bv = bias[n];
#pragma unroll
    for (int r = 0; r < 8; ++r) {
      int row = m0 + r + mb;
      C[(size_t)row * N + n] = acc[j][r] + bv;
    }
  }
}

__device__ __forceinline__ float sigm(float x) {
  return 1.0f / (1.0f + __expf(-x));
}
__device__ __forceinline__ float tanh_fast(float x) {
  float e = __expf(-2.0f * x);
  return (1.0f - e) / (1.0f + e);
}

// Persistent scan: 32 blocks = 4 batch-groups x 8 column-slices.
// h(t-1) tile staged each step by one TDM descriptor per block.
__global__ void gru_scan_wmma(const float* __restrict__ igates,        // BT x 3H fp32
                              const __hip_bfloat16* __restrict__ whh,  // 3H x H bf16
                              const float* __restrict__ b_n,           // H
                              __hip_bfloat16* __restrict__ hstate,     // B x H bf16
                              __hip_bfloat16* __restrict__ houts,      // B x T x H bf16
                              float* __restrict__ hfinal,              // B x H fp32
                              unsigned* __restrict__ counter) {
  __shared__ __align__(16) __hip_bfloat16 hsh[32 * 520];  // 16 rows, TDM-padded
  const int tid = threadIdx.x;
  const int lane = tid & 31;
  const int wave = tid >> 5;
  const int bg = blockIdx.x >> 3;        // batch group 0..3
  const int cs = blockIdx.x & 7;         // column slice 0..7
  const int rowb = bg * 16;
  const int hcol0 = cs * 128 + wave * 16;
  const int nL = lane & 15;
  const unsigned lds_base = (unsigned)(uintptr_t)(&hsh[0]);
  const __hip_bfloat16* htile = hstate + (size_t)rowb * H_;

  for (int t = 0; t < T_; ++t) {
    // One TDM descriptor pulls the 16 x 1024 bf16 h(t-1) tile into LDS.
    if (wave == 0) {
      tdm_load_tile16(lds_base, htile, H_);
      __builtin_amdgcn_s_wait_tensorcnt(0);
    }
    __syncthreads();

    v8f cr = {}, cz = {}, cn = {};
    for (int k0 = 0; k0 < H_; k0 += 32) {
      v16bf a = frag_a_hsh(hsh, k0, lane);
      v16bf br = frag_b(whh, H_, 0 * H_ + hcol0 + nL, k0, lane);
      v16bf bz = frag_b(whh, H_, 1 * H_ + hcol0 + nL, k0, lane);
      v16bf bn = frag_b(whh, H_, 2 * H_ + hcol0 + nL, k0, lane);
      cr = __builtin_amdgcn_wmma_f32_16x16x32_bf16(false, a, false, br, (short)0, cr,
                                                   false, false);
      cz = __builtin_amdgcn_wmma_f32_16x16x32_bf16(false, a, false, bz, (short)0, cz,
                                                   false, false);
      cn = __builtin_amdgcn_wmma_f32_16x16x32_bf16(false, a, false, bn, (short)0, cn,
                                                   false, false);
    }

    const int hcol = hcol0 + nL;
    const float bnv = b_n[hcol];
    const int mb = (lane >> 4) << 3;
#pragma unroll
    for (int r = 0; r < 8; ++r) {
      int m = r + mb;
      int brow = rowb + m;
      const float* ig = igates + ((size_t)brow * T_ + t) * (size_t)G_;
      float ir = ig[hcol];
      float iz = ig[H_ + hcol];
      float in_ = ig[2 * H_ + hcol];
      float hp = __bfloat162float(hsh[hoff(m, hcol)]);
      float rg = sigm(ir + cr[r]);
      float zg = sigm(iz + cz[r]);
      float ng = tanh_fast(in_ + rg * (cn[r] + bnv));
      float hn = (1.0f - zg) * ng + zg * hp;
      __hip_bfloat16 hb = __float2bfloat16(hn);
      hstate[(size_t)brow * H_ + hcol] = hb;
      houts[((size_t)brow * T_ + t) * (size_t)H_ + hcol] = hb;
      if (t == T_ - 1) hfinal[(size_t)brow * H_ + hcol] = hn;
      if (t + 1 < T_) __builtin_prefetch(ig + G_ + hcol, 0, 1);  // next-step igates (HBM)
    }

    // Grid-wide barrier between timesteps (monotonic target => replay-safe).
    __threadfence();
    __syncthreads();
    if (tid == 0) {
      atomicAdd(counter, 1u);
      const unsigned target = (unsigned)(NBLK * (t + 1));
      while (__hip_atomic_load(counter, __ATOMIC_ACQUIRE, __HIP_MEMORY_SCOPE_AGENT) <
             target) {
        __builtin_amdgcn_s_sleep(1);
      }
    }
    __syncthreads();
  }
}

extern "C" void kernel_launch(void* const* d_in, const int* in_sizes, int n_in,
                              void* d_out, int out_size, void* d_ws, size_t ws_size,
                              hipStream_t stream) {
  const float* input = (const float*)d_in[0];   // B,T,I
  const float* w_ih  = (const float*)d_in[1];   // 3H,I
  const float* w_hh  = (const float*)d_in[2];   // 3H,H
  const float* b     = (const float*)d_in[3];   // 3H
  const float* b_n   = (const float*)d_in[4];   // H
  const float* w_lin = (const float*)d_in[5];   // O,H
  const float* b_lin = (const float*)d_in[6];   // O

  char* ws = (char*)d_ws;
  auto alloc = [&](size_t bytes) {
    char* p = ws;
    ws += (bytes + 255) & ~(size_t)255;
    return p;
  };
  __hip_bfloat16* in_bf   = (__hip_bfloat16*)alloc((size_t)BT_ * I_ * 2);
  __hip_bfloat16* wih_bf  = (__hip_bfloat16*)alloc((size_t)G_ * I_ * 2);
  __hip_bfloat16* whh_bf  = (__hip_bfloat16*)alloc((size_t)G_ * H_ * 2);
  __hip_bfloat16* wlin_bf = (__hip_bfloat16*)alloc((size_t)O_ * H_ * 2);
  float*          igates  = (float*)alloc((size_t)BT_ * G_ * 4);
  __hip_bfloat16* houts   = (__hip_bfloat16*)alloc((size_t)BT_ * H_ * 2);
  __hip_bfloat16* hstate  = (__hip_bfloat16*)alloc((size_t)B_ * H_ * 2);
  unsigned*       counter = (unsigned*)alloc(256);

  float* out_h  = (float*)d_out;                 // (B,H) final hidden
  float* out_ys = out_h + (size_t)B_ * H_;       // (B,T,O)

  cast_f32_bf16<<<2048, 256, 0, stream>>>(input, in_bf, BT_ * I_);
  cast_f32_bf16<<<1024, 256, 0, stream>>>(w_ih, wih_bf, G_ * I_);
  cast_f32_bf16<<<1024, 256, 0, stream>>>(w_hh, whh_bf, G_ * H_);
  cast_f32_bf16<<<512, 256, 0, stream>>>(w_lin, wlin_bf, O_ * H_);
  init_scan<<<(B_ * H_ + 255) / 256, 256, 0, stream>>>(hstate, counter);

  // igates = input @ w_ih^T + b  (M=BT, N=3H, K=I)  -- TDM-staged A tiles
  dim3 g1(G_ / 512, BT_ / 16);
  gemm_bias_wmma<<<g1, 256, 0, stream>>>(in_bf, wih_bf, b, igates, BT_, G_, I_);

  // sequential GRU scan (TDM-staged h tiles)
  gru_scan_wmma<<<NBLK, 256, 0, stream>>>(igates, whh_bf, b_n, hstate, houts, out_h,
                                          counter);

  // out_ys = houts @ w_lin^T + b_lin  (M=BT, N=O, K=H) -- async-LDS staged A tiles
  dim3 g2(O_ / 512, BT_ / 16);
  gemm_bias_wmma<<<g2, 256, 0, stream>>>(houts, wlin_bf, b_lin, out_ys, BT_, O_, H_);
}
